// RegionProposalNetwork_78134045049426
// MI455X (gfx1250) — matrix operand order repaired and checked by reading, not compiled, not run on hardware
//
#include <hip/hip_runtime.h>
#include <hip/hip_bf16.h>

typedef __attribute__((ext_vector_type(16))) __bf16 v16bf;
typedef __attribute__((ext_vector_type(8)))  float  v8f;

#define N_IMG   8
#define CCH     512
#define HH      50
#define WW      63
#define HWS     (HH*WW)        /* 3150  */
#define MROWS   (N_IMG*HWS)    /* 25200 */
#define NA      9
#define ANCH    (HWS*NA)       /* 28350 */
#define PRE_NMS 6000
#define POST_NMS 300
#define SORTN   32768
#define NEGC    (-1e30f)

/* d_out layout (float elements), concatenated in reference return order */
#define O_LOCS   0
#define O_SCORES 907200
#define O_ROIS   1360800
#define O_IDX    1370400
#define O_ANCHOR 1372800

__device__ __forceinline__ unsigned short f2bf(float f) {
    unsigned int u = __float_as_uint(f);
    u += 0x7FFFu + ((u >> 16) & 1u);   // round-to-nearest-even
    return (unsigned short)(u >> 16);
}

/* fragment unions: 16 bf16 = 2 x 16B */
union FragA { v16bf v; uint4 q[2]; };

/* ------------------------------------------------------------------ */
/* x: NCHW f32 -> NHWC bf16  (makes GEMM-K contiguous)                 */
/* ------------------------------------------------------------------ */
__global__ void k_cvt_x(const float* __restrict__ x, unsigned short* __restrict__ xb) {
    int i = blockIdx.x * blockDim.x + threadIdx.x;    // over N*HWS*CCH, c fastest
    if (i >= N_IMG * HWS * CCH) return;
    int c = i % CCH; int t = i / CCH;
    int s = t % HWS; int n = t / HWS;
    xb[i] = f2bf(x[((size_t)(n * CCH + c)) * HWS + s]);
}

/* conv1_w (O,C,3,3) -> wb2[tap][o][c] bf16 (k=c contiguous per output col) */
__global__ void k_pack_w(const float* __restrict__ w, unsigned short* __restrict__ wb2) {
    int i = blockIdx.x * blockDim.x + threadIdx.x;    // over 9*512*512, c fastest
    if (i >= 9 * CCH * CCH) return;
    int c = i % CCH; int t = i / CCH;
    int o = t % CCH; int tap = t / CCH;
    int ky = tap / 3, kx = tap % 3;
    wb2[i] = f2bf(w[((o * CCH + c) * 3 + ky) * 3 + kx]);
}

/* head weights packed [n][k]: n=0..35 loc rows, 36..53 score rows, 54..63 pad */
__global__ void k_pack_head(const float* __restrict__ lw, const float* __restrict__ lb,
                            const float* __restrict__ sw, const float* __restrict__ sb,
                            unsigned short* __restrict__ hw2, float* __restrict__ hb) {
    int i = blockIdx.x * blockDim.x + threadIdx.x;    // 64*512, k fastest
    if (i >= 64 * CCH) return;
    int k = i % CCH, n = i / CCH;
    float v = 0.f;
    if (n < 36)      v = lw[n * CCH + k];
    else if (n < 54) v = sw[(n - 36) * CCH + k];
    hw2[i] = f2bf(v);
    if (k == 0) hb[n] = (n < 36) ? lb[n] : ((n < 54) ? sb[n - 36] : 0.f);
}

/* ------------------------------------------------------------------ */
/* 3x3 conv implicit GEMM: block 256 thr = 8 waves -> 64(M) x 128(N)   */
/* each wave: 2x2 16x16 tiles = 4 WMMAs / 32-K chunk; K = 9*512        */
/* ------------------------------------------------------------------ */
__global__ void __launch_bounds__(256) k_conv_wmma(
        const unsigned short* __restrict__ xb,     /* NHWC bf16 */
        const unsigned short* __restrict__ wb2,    /* [tap][o][c] bf16 */
        const float* __restrict__ bias,
        unsigned short* __restrict__ h16) {        /* [m][o] bf16 */
    __shared__ __align__(16) unsigned short As[64][40];   /* [row][k]  */
    __shared__ __align__(16) unsigned short Bt[128][40];  /* [col][k]  */
    const int tid  = threadIdx.x;
    const int lane = tid & 31;
    const int wave = tid >> 5;
    const int wr   = wave >> 2;        /* 0..1 : M half  */
    const int wc   = wave & 3;         /* 0..3 : N quarter */
    const int m0   = blockIdx.x * 64;
    const int n0   = blockIdx.y * 128;

    /* per-thread A staging coords: row r = tid>>2, k-group = (tid&3)*8 */
    const int sar = tid >> 2;
    const int sak = (tid & 3) * 8;
    int aimg = 0, ayy = 0, axx = 0; bool amv;
    {
        int m = m0 + sar;
        amv = (m < MROWS);
        int mm = amv ? m : 0;
        aimg = mm / HWS;
        int s = mm % HWS;
        ayy = s / WW; axx = s % WW;
    }
    /* per-thread B staging coords: col = tid>>1, k-group = (tid&1)*16 */
    const int sbc = tid >> 1;
    const int sbk = (tid & 1) * 16;

    v8f acc00 = {0,0,0,0,0,0,0,0}, acc01 = acc00, acc10 = acc00, acc11 = acc00;

    const int l15  = lane & 15;
    const int klo  = (lane < 16) ? 0 : 8;    /* A k-offset for hi lanes  */
    const int kbhi = (lane < 16) ? 0 : 16;   /* unused; B reads use klo too  */
    const int ar0  = wr * 32 + l15;          /* A rows for the two tiles */
    const int ar1  = ar0 + 16;
    const int bc0  = wc * 32 + l15;          /* B cols for the two tiles */
    const int bc1  = bc0 + 16;

    const uint4 zq = {0u, 0u, 0u, 0u};

    for (int tap = 0; tap < 9; ++tap) {
        const int dy = tap / 3 - 1, dx = tap % 3 - 1;
        const int yy = ayy + dy, xx = axx + dx;
        const bool av = amv && yy >= 0 && yy < HH && xx >= 0 && xx < WW;
        const unsigned short* abase =
            xb + ((size_t)((aimg * HH + yy) * WW + xx)) * CCH + sak;
        const unsigned short* bbase =
            wb2 + ((size_t)(tap * CCH + n0 + sbc)) * CCH + sbk;

        for (int kc = 0; kc < 16; ++kc) {
            const int cb = kc * 32;
            /* stage A (1 x b128 per thread) */
            uint4 aq = av ? *(const uint4*)(abase + cb) : zq;
            *(uint4*)&As[sar][sak] = aq;
            /* stage B transposed (2 x b128 per thread) */
            uint4 bq0 = *(const uint4*)(bbase + cb);
            uint4 bq1 = *(const uint4*)(bbase + cb + 8);
            *(uint4*)&Bt[sbc][sbk]     = bq0;
            *(uint4*)&Bt[sbc][sbk + 8] = bq1;
            if (tid == 0 && (kc + 1) < 16)
                __builtin_prefetch(bbase + cb + 32, 0, 1);
            __syncthreads();

            /* fragments: two contiguous 8-elem k-runs each -> 2 ds_load_b128 */
            FragA a0, a1, b0, b1;
            a0.q[0] = *(const uint4*)&As[ar0][klo];
            a0.q[1] = *(const uint4*)&As[ar0][16 + klo];
            a1.q[0] = *(const uint4*)&As[ar1][klo];
            a1.q[1] = *(const uint4*)&As[ar1][16 + klo];
            b0.q[0] = *(const uint4*)&Bt[bc0][klo];
            b0.q[1] = *(const uint4*)&Bt[bc0][16 + klo];
            b1.q[0] = *(const uint4*)&Bt[bc1][klo];
            b1.q[1] = *(const uint4*)&Bt[bc1][16 + klo];

            acc00 = __builtin_amdgcn_wmma_f32_16x16x32_bf16(false, a0.v, false, b0.v, (short)0, acc00, false, false);
            acc01 = __builtin_amdgcn_wmma_f32_16x16x32_bf16(false, a0.v, false, b1.v, (short)0, acc01, false, false);
            acc10 = __builtin_amdgcn_wmma_f32_16x16x32_bf16(false, a1.v, false, b0.v, (short)0, acc10, false, false);
            acc11 = __builtin_amdgcn_wmma_f32_16x16x32_bf16(false, a1.v, false, b1.v, (short)0, acc11, false, false);
            __syncthreads();
        }
    }

    /* bias + ReLU, store h bf16 in [m][o] layout */
    const int rbase = (lane < 16) ? 0 : 8;
    const int col0 = n0 + wc * 32 + l15;
    const int col1 = col0 + 16;
    const float bv0 = bias[col0], bv1 = bias[col1];
#pragma unroll
    for (int v = 0; v < 8; ++v) {
        int mA = m0 + wr * 32 + rbase + v;
        int mB = mA + 16;
        if (mA < MROWS) {
            float v00 = fmaxf(acc00[v] + bv0, 0.f);
            float v01 = fmaxf(acc01[v] + bv1, 0.f);
            h16[(size_t)mA * CCH + col0] = f2bf(v00);
            h16[(size_t)mA * CCH + col1] = f2bf(v01);
        }
        if (mB < MROWS) {
            float v10 = fmaxf(acc10[v] + bv0, 0.f);
            float v11 = fmaxf(acc11[v] + bv1, 0.f);
            h16[(size_t)mB * CCH + col0] = f2bf(v10);
            h16[(size_t)mB * CCH + col1] = f2bf(v11);
        }
    }
}

/* ------------------------------------------------------------------ */
/* head GEMM: M=25200, N=64 (36 loc | 18 score | 10 pad), K=512        */
/* block 256 thr = 8 waves -> 64(M) x 64(N); wave: 2 M-tiles, 1 N-tile */
/* ------------------------------------------------------------------ */
__global__ void __launch_bounds__(256) k_head_wmma(
        const unsigned short* __restrict__ h16,    /* [m][k] bf16 */
        const unsigned short* __restrict__ hw2,    /* [n][k] bf16 */
        const float* __restrict__ hb,
        float* __restrict__ out) {
    __shared__ __align__(16) unsigned short As[64][40];
    __shared__ __align__(16) unsigned short Bt[64][40];
    const int tid  = threadIdx.x;
    const int lane = tid & 31;
    const int wave = tid >> 5;
    const int wr   = wave >> 2;        /* 0..1 */
    const int wc   = wave & 3;         /* 0..3 -> N tile */
    const int m0   = blockIdx.x * 64;

    const int sar = tid >> 2;            /* A: row, kg=(tid&3)*8       */
    const int sak = (tid & 3) * 8;
    const int sbc = tid >> 2;            /* B: col=tid>>2, kg=(tid&3)*8 */
    const int sbk = (tid & 3) * 8;
    const bool amv = (m0 + sar) < MROWS;
    const unsigned short* abase = h16 + (size_t)(amv ? (m0 + sar) : 0) * CCH + sak;
    const unsigned short* bbase = hw2 + (size_t)sbc * CCH + sbk;

    v8f acc0 = {0,0,0,0,0,0,0,0}, acc1 = acc0;
    const int l15  = lane & 15;
    const int klo  = (lane < 16) ? 0 : 8;
    const int kbhi = (lane < 16) ? 0 : 16;
    const int ar0  = wr * 32 + l15;
    const int ar1  = ar0 + 16;
    const int bc   = wc * 16 + l15;
    const uint4 zq = {0u, 0u, 0u, 0u};

    for (int kc = 0; kc < 16; ++kc) {
        const int cb = kc * 32;
        uint4 aq = amv ? *(const uint4*)(abase + cb) : zq;
        *(uint4*)&As[sar][sak] = aq;
        *(uint4*)&Bt[sbc][sbk] = *(const uint4*)(bbase + cb);
        __syncthreads();
        FragA a0, a1, b0;
        a0.q[0] = *(const uint4*)&As[ar0][klo];
        a0.q[1] = *(const uint4*)&As[ar0][16 + klo];
        a1.q[0] = *(const uint4*)&As[ar1][klo];
        a1.q[1] = *(const uint4*)&As[ar1][16 + klo];
        b0.q[0] = *(const uint4*)&Bt[bc][klo];
        b0.q[1] = *(const uint4*)&Bt[bc][16 + klo];
        acc0 = __builtin_amdgcn_wmma_f32_16x16x32_bf16(false, a0.v, false, b0.v, (short)0, acc0, false, false);
        acc1 = __builtin_amdgcn_wmma_f32_16x16x32_bf16(false, a1.v, false, b0.v, (short)0, acc1, false, false);
        __syncthreads();
    }

    const int nout  = wc * 16 + l15;
    const int rbase = (lane < 16) ? 0 : 8;
    const float bv  = hb[nout];
#pragma unroll
    for (int v = 0; v < 8; ++v) {
#pragma unroll
        for (int ta = 0; ta < 2; ++ta) {
            int m = m0 + wr * 32 + ta * 16 + rbase + v;
            if (m >= MROWS) continue;
            float val = (ta ? acc1[v] : acc0[v]) + bv;
            int img = m / HWS, s = m % HWS;
            if (nout < 36) {
                int a = nout >> 2, d = nout & 3;
                out[O_LOCS + ((size_t)img * ANCH + (size_t)s * NA + a) * 4 + d] = val;
            } else if (nout < 54) {
                int j = nout - 36; int a = j >> 1, cls = j & 1;
                out[O_SCORES + ((size_t)img * ANCH + (size_t)s * NA + a) * 2 + cls] = val;
            }
        }
    }
}

/* ------------------------------------------------------------------ */
/* anchors */
/* ------------------------------------------------------------------ */
__global__ void k_anchor(float* __restrict__ out) {
    int i = blockIdx.x * blockDim.x + threadIdx.x;
    if (i >= ANCH) return;
    int a = i % NA, s = i / NA;
    int y = s / WW, x = s % WW;
    const float ratios[3] = {0.5f, 1.f, 2.f};
    const float scales[3] = {8.f, 16.f, 32.f};
    float r = ratios[a / 3], sc = scales[a % 3];
    float h = 16.f * sc * sqrtf(r);
    float w = 16.f * sc * sqrtf(1.f / r);
    float py = 8.f + y * 16.f, px = 8.f + x * 16.f;
    float* o = out + O_ANCHOR + (size_t)i * 4;
    o[0] = py - 0.5f * h; o[1] = px - 0.5f * w;
    o[2] = py + 0.5f * h; o[3] = px + 0.5f * w;
}

/* ------------------------------------------------------------------ */
/* decode + clip + min-size filter; fill padded (score,idx) sort arrays */
/* ------------------------------------------------------------------ */
__global__ void k_decode(const float* __restrict__ out_ro,
                         const int* __restrict__ ihp, const int* __restrict__ iwp,
                         float* __restrict__ roi,
                         float* __restrict__ scp, int* __restrict__ idp) {
    int t = blockIdx.x * blockDim.x + threadIdx.x;
    int img = t >> 15, j = t & (SORTN - 1);
    if (img >= N_IMG) return;
    if (j < ANCH) {
        const float* an = out_ro + O_ANCHOR + (size_t)j * 4;
        const float* lc = out_ro + O_LOCS + ((size_t)img * ANCH + j) * 4;
        float ah = an[2] - an[0], aw = an[3] - an[1];
        float cy = an[0] + 0.5f * ah, cx = an[1] + 0.5f * aw;
        float cy2 = lc[0] * ah + cy, cx2 = lc[1] * aw + cx;
        float h2 = expf(lc[2]) * ah, w2 = expf(lc[3]) * aw;
        float fh = (float)(*ihp), fw = (float)(*iwp);
        float y1 = fminf(fmaxf(cy2 - 0.5f * h2, 0.f), fh);
        float x1 = fminf(fmaxf(cx2 - 0.5f * w2, 0.f), fw);
        float y2 = fminf(fmaxf(cy2 + 0.5f * h2, 0.f), fh);
        float x2 = fminf(fmaxf(cx2 + 0.5f * w2, 0.f), fw);
        float* rp = roi + ((size_t)img * ANCH + j) * 4;
        rp[0] = y1; rp[1] = x1; rp[2] = y2; rp[3] = x2;
        bool keep = (y2 - y1 >= 16.f) && (x2 - x1 >= 16.f);
        float fg = out_ro[O_SCORES + ((size_t)img * ANCH + j) * 2 + 1];
        scp[(size_t)img * SORTN + j] = keep ? fg : NEGC;
    } else {
        scp[(size_t)img * SORTN + j] = NEGC;
    }
    idp[(size_t)img * SORTN + j] = j;
}

/* ------------------------------------------------------------------ */
/* global bitonic sort pass (descending, index-stable)                 */
/* ------------------------------------------------------------------ */
__global__ void k_bitonic(float* __restrict__ sc, int* __restrict__ id, int k, int j) {
    int t = blockIdx.x * blockDim.x + threadIdx.x;   // 8 * 16384
    int img = t >> 14, l = t & 16383;
    if (img >= N_IMG) return;
    int low = l & (j - 1);
    int i = ((l ^ low) << 1) | low;
    int p = i + j;
    float* s = sc + (size_t)img * SORTN;
    int*   d = id + (size_t)img * SORTN;
    bool desc = ((i & k) == 0);
    float si = s[i], sp = s[p];
    int   di = d[i], dp = d[p];
    bool wrong = desc ? ((si < sp) || (si == sp && di > dp))
                      : ((si > sp) || (si == sp && di < dp));
    if (wrong) { s[i] = sp; s[p] = si; d[i] = dp; d[p] = di; }
}

/* ------------------------------------------------------------------ */
/* greedy NMS: 1 block per image, serial 300 rounds of argmax+suppress */
/* ------------------------------------------------------------------ */
__global__ void __launch_bounds__(256) k_nms(
        const float* __restrict__ scp, const int* __restrict__ idp,
        const float* __restrict__ roi, float* __restrict__ topbox,
        float* __restrict__ out) {
    __shared__ float sc_s[PRE_NMS];
    __shared__ float area_s[PRE_NMS];
    __shared__ float redv[256];
    __shared__ int   redi[256];
    __shared__ int   s_pick;
    __shared__ int   s_valid;
    const int img = blockIdx.x, tid = threadIdx.x;

    for (int jj = tid; jj < PRE_NMS; jj += 256) {
        float s = scp[(size_t)img * SORTN + jj];
        int  si = idp[(size_t)img * SORTN + jj];
        float b0 = 0.f, b1 = 0.f, b2 = 0.f, b3 = 0.f;
        if (si < ANCH) {
            const float* rp = roi + ((size_t)img * ANCH + si) * 4;
            b0 = rp[0]; b1 = rp[1]; b2 = rp[2]; b3 = rp[3];
        }
        float* tb = topbox + ((size_t)img * PRE_NMS + jj) * 4;
        tb[0] = b0; tb[1] = b1; tb[2] = b2; tb[3] = b3;
        sc_s[jj]   = s;
        area_s[jj] = (b2 - b0) * (b3 - b1);
    }
    __syncthreads();

    for (int it = 0; it < POST_NMS; ++it) {
        float best = -3.0e38f; int bi = PRE_NMS;
        for (int jj = tid; jj < PRE_NMS; jj += 256) {
            float s = sc_s[jj];
            if (s > best || (s == best && jj < bi)) { best = s; bi = jj; }
        }
        redv[tid] = best; redi[tid] = bi;
        __syncthreads();
        for (int off = 128; off > 0; off >>= 1) {
            if (tid < off) {
                float ov = redv[tid + off]; int oi = redi[tid + off];
                if (ov > redv[tid] || (ov == redv[tid] && oi < redi[tid])) {
                    redv[tid] = ov; redi[tid] = oi;
                }
            }
            __syncthreads();
        }
        if (tid == 0) { s_pick = redi[0]; s_valid = (redv[0] > NEGC * 0.5f) ? 1 : 0; }
        __syncthreads();
        const int pi = s_pick;
        const float* pb = topbox + ((size_t)img * PRE_NMS + pi) * 4;
        float p0 = pb[0], p1 = pb[1], p2 = pb[2], p3 = pb[3];
        float pa = area_s[pi];
        for (int jj = tid; jj < PRE_NMS; jj += 256) {
            const float* qb = topbox + ((size_t)img * PRE_NMS + jj) * 4;
            float ty = fmaxf(p0, qb[0]), tx = fmaxf(p1, qb[1]);
            float by = fminf(p2, qb[2]), bx = fminf(p3, qb[3]);
            float w0 = fmaxf(by - ty, 0.f), w1 = fmaxf(bx - tx, 0.f);
            float inter = w0 * w1;
            float iou = inter / (pa + area_s[jj] - inter + 1e-9f);
            if (iou > 0.7f) sc_s[jj] = NEGC;
        }
        if (tid == 0) {
            int r = img * POST_NMS + it;
            float* ro = out + O_ROIS + (size_t)r * 4;
            if (s_valid) { ro[0] = p0; ro[1] = p1; ro[2] = p2; ro[3] = p3; }
            else         { ro[0] = 0.f; ro[1] = 0.f; ro[2] = 0.f; ro[3] = 0.f; }
            ((int*)(out + O_IDX))[r] = img;
        }
        __syncthreads();
    }
}

/* ------------------------------------------------------------------ */
extern "C" void kernel_launch(void* const* d_in, const int* in_sizes, int n_in,
                              void* d_out, int out_size, void* d_ws, size_t ws_size,
                              hipStream_t stream) {
    (void)in_sizes; (void)n_in; (void)out_size; (void)ws_size;
    const float* x   = (const float*)d_in[0];
    const float* w1  = (const float*)d_in[1];
    const float* b1  = (const float*)d_in[2];
    const float* lw  = (const float*)d_in[3];
    const float* lb  = (const float*)d_in[4];
    const float* sw  = (const float*)d_in[5];
    const float* sb  = (const float*)d_in[6];
    const int*   ihp = (const int*)d_in[7];
    const int*   iwp = (const int*)d_in[8];
    float* out = (float*)d_out;

    char* ws = (char*)d_ws;
    size_t off = 0;
    auto carve = [&](size_t bytes) -> void* {
        void* p = ws + off;
        off += (bytes + 255) & ~(size_t)255;
        return p;
    };
    unsigned short* xb  = (unsigned short*)carve((size_t)N_IMG * HWS * CCH * 2);
    unsigned short* wb2 = (unsigned short*)carve((size_t)9 * CCH * CCH * 2);
    unsigned short* h16 = (unsigned short*)carve((size_t)MROWS * CCH * 2);
    unsigned short* hw2 = (unsigned short*)carve((size_t)64 * CCH * 2);
    float*          hbp = (float*)carve(64 * 4);
    float*          roi = (float*)carve((size_t)N_IMG * ANCH * 4 * 4);
    float*          scp = (float*)carve((size_t)N_IMG * SORTN * 4);
    int*            idp = (int*)carve((size_t)N_IMG * SORTN * 4);
    float*          tbx = (float*)carve((size_t)N_IMG * PRE_NMS * 4 * 4);

    const int nx = N_IMG * HWS * CCH;
    k_cvt_x<<<(nx + 255) / 256, 256, 0, stream>>>(x, xb);
    k_pack_w<<<(9 * CCH * CCH + 255) / 256, 256, 0, stream>>>(w1, wb2);
    k_pack_head<<<(64 * CCH + 255) / 256, 256, 0, stream>>>(lw, lb, sw, sb, hw2, hbp);
    k_anchor<<<(ANCH + 255) / 256, 256, 0, stream>>>(out);

    dim3 gconv((MROWS + 63) / 64, CCH / 128);
    k_conv_wmma<<<gconv, 256, 0, stream>>>(xb, wb2, b1, h16);
    k_head_wmma<<<(MROWS + 63) / 64, 256, 0, stream>>>(h16, hw2, hbp, out);

    k_decode<<<(N_IMG * SORTN) / 256, 256, 0, stream>>>(out, ihp, iwp, roi, scp, idp);

    for (int k = 2; k <= SORTN; k <<= 1)
        for (int j = k >> 1; j >= 1; j >>= 1)
            k_bitonic<<<(N_IMG * (SORTN / 2)) / 256, 256, 0, stream>>>(scp, idp, k, j);

    k_nms<<<N_IMG, 256, 0, stream>>>(scp, idp, roi, tbx, out);
}
